// BiGRUEncoder_41970420416938
// MI455X (gfx1250) — compile-verified, matrix operand order checked by reading, not run on hardware
//
#include <hip/hip_runtime.h>
#include <hip/hip_bf16.h>

typedef __bf16 v16bf __attribute__((ext_vector_type(16)));
typedef float  v8f   __attribute__((ext_vector_type(8)));

#define Bk 16
#define Tk 2048
#define Nk 1025
#define Fk 512
#define G3 1536
#define NBLK 16     // blocks per direction in recurrent phase
#define COLS 32     // hidden columns per block
#define NROWS 96    // gate rows (3 gates x 32 cols) per block

// ---------------------------------------------------------------------------
// Phase 1: gi[dir][t][b][:] = x[t or T-1-t][b] @ Wih^T   (bf16 WMMA, f32 acc)
// grid = (12 n-blocks of 128 cols, T, 2 dirs), block = 256 (8 waves x 16 cols)
// ---------------------------------------------------------------------------
__global__ __launch_bounds__(256) void gru_input_proj(
    const float* __restrict__ x,       // (B,T,N)
    const float* __restrict__ Wih_f,   // (3F,F)
    const float* __restrict__ Wih_b,
    __bf16* __restrict__ gi)           // (2,T,B,3F) bf16
{
  const int dir  = blockIdx.z;
  const int t    = blockIdx.y;
  const int tsrc = dir ? (Tk - 1 - t) : t;
  const int lane = threadIdx.x & 31;
  const int wave = threadIdx.x >> 5;
  const int half = lane >> 4;
  const int m    = lane & 15;                    // batch row (A) / n col (B)
  const int nbase = blockIdx.x * 128 + wave * 16;
  const float* W = dir ? Wih_b : Wih_f;

  const float* arow = x + ((size_t)m * Tk + tsrc) * Nk;   // x row for batch m
  const float* brow = W + (size_t)(nbase + m) * Fk;       // Wih row (= B col)

  v8f acc = {0.f, 0.f, 0.f, 0.f, 0.f, 0.f, 0.f, 0.f};
  for (int kb = 0; kb < Fk; kb += 32) {
    v16bf a, b;
    const int k0 = kb + half * 8;
    #pragma unroll
    for (int e = 0; e < 8; ++e) {
      a[e]     = (__bf16)arow[k0 + e];
      a[e + 8] = (__bf16)arow[k0 + 16 + e];
      b[e]     = (__bf16)brow[k0 + e];
      b[e + 8] = (__bf16)brow[k0 + 16 + e];
    }
    acc = __builtin_amdgcn_wmma_f32_16x16x32_bf16(false, a, false, b,
                                                  (short)0, acc, false, false);
  }
  // C layout: VGPR v -> row M = v + 8*half, col N = lane&15
  __bf16* outp = gi + (((size_t)dir * Tk + t) * Bk) * G3 + nbase + m;
  #pragma unroll
  for (int v = 0; v < 8; ++v) {
    outp[(size_t)(v + 8 * half) * G3] = (__bf16)acc[v];
  }
}

// ---------------------------------------------------------------------------
// Phase 2: sequential scan. grid = (16 col-slices, 2 dirs), block = 192 (6 waves).
// Whh slice lives in LDS (bf16) for the whole scan; one grid barrier per step.
// ---------------------------------------------------------------------------
__global__ __launch_bounds__(192) void gru_recurrent(
    const float* __restrict__ x,
    const float* __restrict__ Whh_f, const float* __restrict__ Whh_b,
    const float* __restrict__ bih_f, const float* __restrict__ bhh_f,
    const float* __restrict__ bih_b, const float* __restrict__ bhh_b,
    const __bf16* __restrict__ gi,     // (2,T,B,3F)
    float* __restrict__ hstate,        // (2 dirs, 2 bufs, B, F) f32, pre-zeroed
    unsigned* __restrict__ cnt,        // (2, T) u32, pre-zeroed
    const int* __restrict__ Lp,
    float* __restrict__ out)           // (B, T-2L, 2F)
{
  __shared__ __bf16 w_lds[NROWS][Fk];   // 96 KB
  __shared__ __bf16 h_lds[Bk][Fk];      // 16 KB
  __shared__ float  gh_lds[Bk][NROWS];  //  6 KB

  const int dir   = blockIdx.y;
  const int cbase = blockIdx.x * COLS;
  const int tid   = threadIdx.x;
  const int lane  = tid & 31;
  const int wave  = tid >> 5;
  const int half  = lane >> 4;
  const int m     = lane & 15;
  const int Lv    = Lp[0];
  const int Tp    = Tk - 2 * Lv;

  const float* Whh = dir ? Whh_b : Whh_f;
  const float* bih = dir ? bih_b : bih_f;
  const float* bhh = dir ? bhh_b : bhh_f;
  float*    hbuf  = hstate + (size_t)dir * 2 * Bk * Fk;
  unsigned* mycnt = cnt + (size_t)dir * Tk;
  const __bf16* gid = gi + ((size_t)dir * Tk) * Bk * G3;

  // Load resident Whh slice: n in [0,96) -> Whh row g*F + cbase + c
  for (int i = tid; i < NROWS * Fk; i += blockDim.x) {
    int n = i >> 9, k = i & 511;
    int g = n >> 5, c = n & 31;
    w_lds[n][k] = (__bf16)Whh[(size_t)(g * Fk + cbase + c) * Fk + k];
  }
  __syncthreads();

  for (int t = 0; t < Tk; ++t) {
    if (t > 0) {
      if (tid == 0) {
        while (__hip_atomic_load(&mycnt[t - 1], __ATOMIC_ACQUIRE,
                                 __HIP_MEMORY_SCOPE_AGENT) < NBLK) { }
      }
      __syncthreads();
    }
    const float* hprev = hbuf + (size_t)(t & 1) * Bk * Fk;
    float*       hnext = hbuf + (size_t)((t + 1) & 1) * Bk * Fk;

    // stage full h_prev as bf16
    for (int i = tid; i < Bk * Fk; i += blockDim.x)
      h_lds[i >> 9][i & 511] = (__bf16)hprev[i];
    __syncthreads();

    // gh_slice = h @ Whh_slice^T : M=16, N=96 (one 16-col tile per wave), K=512
    {
      v8f acc = {0.f, 0.f, 0.f, 0.f, 0.f, 0.f, 0.f, 0.f};
      for (int kb = 0; kb < Fk; kb += 32) {
        v16bf a, b;
        const int k0 = kb + half * 8;
        #pragma unroll
        for (int e = 0; e < 8; ++e) {
          a[e]     = h_lds[m][k0 + e];
          a[e + 8] = h_lds[m][k0 + 16 + e];
          b[e]     = w_lds[wave * 16 + m][k0 + e];
          b[e + 8] = w_lds[wave * 16 + m][k0 + 16 + e];
        }
        acc = __builtin_amdgcn_wmma_f32_16x16x32_bf16(false, a, false, b,
                                                      (short)0, acc, false, false);
      }
      #pragma unroll
      for (int v = 0; v < 8; ++v)
        gh_lds[v + 8 * half][wave * 16 + m] = acc[v];
    }
    __syncthreads();

    // fused gates + residual + output write
    const int tsrc = dir ? (Tk - 1 - t) : t;
    const __bf16* git = gid + ((size_t)t * Bk) * G3;
    for (int idx = tid; idx < Bk * COLS; idx += blockDim.x) {
      int b = idx >> 5, c = idx & 31;
      int j = cbase + c;
      const __bf16* gib = git + (size_t)b * G3;
      float ir = (float)gib[j]           + bih[j];
      float iz = (float)gib[Fk + j]      + bih[Fk + j];
      float in = (float)gib[2 * Fk + j]  + bih[2 * Fk + j];
      float hr = gh_lds[b][c]       + bhh[j];
      float hz = gh_lds[b][32 + c]  + bhh[Fk + j];
      float hn = gh_lds[b][64 + c]  + bhh[2 * Fk + j];
      float r = 1.f / (1.f + __expf(-(ir + hr)));
      float z = 1.f / (1.f + __expf(-(iz + hz)));
      float n = tanhf(in + r * hn);
      float hp = hprev[b * Fk + j];
      float xv = x[((size_t)b * Tk + tsrc) * Nk + j];
      float h2 = (1.f - z) * n + z * hp + xv;
      hnext[b * Fk + j] = h2;
      if (t >= Lv && t < Tk - Lv)
        out[((size_t)b * Tp + (t - Lv)) * (2 * Fk) + dir * Fk + j] = h2;
    }
    __syncthreads();
    if (tid == 0) {
      __threadfence();
      __hip_atomic_fetch_add(&mycnt[t], 1u, __ATOMIC_RELEASE,
                             __HIP_MEMORY_SCOPE_AGENT);
    }
  }
}

// ---------------------------------------------------------------------------
extern "C" void kernel_launch(void* const* d_in, const int* in_sizes, int n_in,
                              void* d_out, int out_size, void* d_ws, size_t ws_size,
                              hipStream_t stream) {
  const float* x     = (const float*)d_in[0];
  const float* Wih_f = (const float*)d_in[1];
  const float* Whh_f = (const float*)d_in[2];
  const float* bih_f = (const float*)d_in[3];
  const float* bhh_f = (const float*)d_in[4];
  const float* Wih_b = (const float*)d_in[5];
  const float* Whh_b = (const float*)d_in[6];
  const float* bih_b = (const float*)d_in[7];
  const float* bhh_b = (const float*)d_in[8];
  const int*   Lp    = (const int*)d_in[9];
  float* out = (float*)d_out;

  unsigned char* ws = (unsigned char*)d_ws;
  unsigned* cnt    = (unsigned*)ws;                         // 2*T*4      = 16 KB
  float*    hstate = (float*)(ws + 16384);                  // 2*2*16*512*4 = 128 KB
  __bf16*   gi     = (__bf16*)(ws + 16384 + 131072);        // 2*T*16*1536*2 ~= 192 MB

  // counters + h state must be zero at scan start (deterministic every call)
  hipMemsetAsync(ws, 0, 16384 + 131072, stream);

  dim3 g1(12, Tk, 2);
  gru_input_proj<<<g1, 256, 0, stream>>>(x, Wih_f, Wih_b, gi);

  dim3 g2(NBLK, 2);
  gru_recurrent<<<g2, 192, 0, stream>>>(x, Whh_f, Whh_b, bih_f, bhh_f,
                                        bih_b, bhh_b, gi, hstate, cnt, Lp, out);
}
